// EncoderDecoder_21930103013491
// MI455X (gfx1250) — matrix operand order repaired and checked by reading, not compiled, not run on hardware
//
#include <hip/hip_runtime.h>
#include <hip/hip_bf16.h>

typedef _Float16 half_t;
typedef __attribute__((ext_vector_type(16))) _Float16 v16h;
typedef __attribute__((ext_vector_type(8)))  _Float16 v8h;
typedef __attribute__((ext_vector_type(8)))  float    v8f;

#define NWG 64
#define NTH 256

namespace {
constexpr int BATCH = 256;
constexpr int T_IN  = 168;
constexpr int T_OUT = 24;
constexpr int HID   = 512;
constexpr int G4    = 2048;          // 4*HID
constexpr int KT_H  = HID / 32;      // 16 K-tiles per K=512
// LDS row strides (halves), +8 pad so A-fragment reads hit distinct banks
constexpr int ST0 = HID + 32 + 8;    // enc L0: [h(512) | x(32)] = 552
constexpr int ST1 = 2 * HID + 8;     // enc L1 / dec cell1: [h(512) | x(512)] = 1032
}

__device__ __forceinline__ float sigm(float x) { return 1.0f / (1.0f + __expf(-x)); }
__device__ __forceinline__ float tanh_(float x) {
  x = fminf(fmaxf(x, -15.0f), 15.0f);
  float e = __expf(2.0f * x);
  return (e - 1.0f) / (e + 1.0f);
}

union V16 { v16h v; v8h h[2]; };

// A fragment (16-bit WMMA A layout): lane holds M=lane&15; halves j<8 -> K=k0+8*hi+j,
// j>=8 -> K=k0+16+8*hi+(j-8).  p must point at (row, k0 + 8*hi).
__device__ __forceinline__ v16h load_a_frag(const half_t* p) {
  V16 t;
  t.h[0] = *(const v8h*)(p);
  t.h[1] = *(const v8h*)(p + 16);
  return t.v;
}

// ---- CDNA5 async global->LDS copy (ASYNCcnt) ----
__device__ __forceinline__ void async_cp16(unsigned lds_off, const half_t* g) {
  asm volatile("global_load_async_to_lds_b128 %0, %1, off"
               :: "v"(lds_off), "v"(g) : "memory");
}
__device__ __forceinline__ void async_wait0() {
  asm volatile("s_wait_asynccnt 0x0" ::: "memory");
}

// Loop-variant zero offset: defeats LICM of loop-invariant weight loads (which
// previously caused scratch spills) while keeping the access rooted at the
// kernel-arg pointer so address-space inference still emits GLOBAL loads.
#define FRESH_ZERO_OFF(name) \
  size_t name = 0;           \
  asm volatile("" : "+s"(name))

// Device-wide barrier: monotonic counter, reset by hipMemsetAsync each launch.
__device__ __forceinline__ void grid_barrier(unsigned* cnt, unsigned* epoch) {
  __threadfence();
  __syncthreads();
  if (threadIdx.x == 0) {
    unsigned e = ++(*epoch);
    atomicAdd(cnt, 1u);
    unsigned target = e * (unsigned)gridDim.x;
    while (__hip_atomic_load(cnt, __ATOMIC_ACQUIRE, __HIP_MEMORY_SCOPE_AGENT) < target)
      __builtin_amdgcn_s_sleep(1);
  }
  __syncthreads();
  __threadfence();
}

// Pack fp32 weight [N][K] into WMMA B-fragment order:
// tile = n_t*(K/32)+k_t; lane holds column N = n_t*16+(lane&15);
// half j<8 -> K = k_t*32 + 8*(lane>=16) + j; j>=8 -> K = k_t*32 + 16 + 8*(lane>=16) + (j-8).
__global__ void pack_w(const float* __restrict__ W, half_t* __restrict__ o, int N, int K) {
  int KT = K >> 5;
  int total = N * K;
  for (int e = blockIdx.x * blockDim.x + threadIdx.x; e < total; e += gridDim.x * blockDim.x) {
    int tile = e >> 9;
    int r = e & 511;
    int lane = r >> 4;
    int j = r & 15;
    int n_t = tile / KT, k_t = tile % KT;
    int n = n_t * 16 + (lane & 15);
    int k = k_t * 32 + ((lane >> 4) << 3) + j + ((j >> 3) << 3);
    o[e] = (half_t)W[(size_t)n * K + k];
  }
}

__global__ void cvt_h(const float* __restrict__ s, half_t* __restrict__ d, int n) {
  for (int i = blockIdx.x * blockDim.x + threadIdx.x; i < n; i += gridDim.x * blockDim.x)
    d[i] = (half_t)s[i];
}

#define WMMA_F16(A, B, C) \
  __builtin_amdgcn_wmma_f32_16x16x32_f16(false, (A), false, (B), (short)0, (C), false, false)

// ---------------- Encoder layer 0: z = h@Whh0^T + x_t@Wih0^T ----------------
__global__ __launch_bounds__(NTH)
void enc_layer0(const half_t* __restrict__ x16,
                const half_t* WhhP, const half_t* WihP,
                const float* __restrict__ b_ih, const float* __restrict__ b_hh,
                float* __restrict__ cbuf,
                half_t* __restrict__ h16, half_t* __restrict__ ys0,
                float* __restrict__ zbuf, unsigned* __restrict__ bar) {
  __shared__ half_t smemA[32 * ST0];
  const int tid = threadIdx.x;
  const int lane = tid & 31;
  const int m = lane & 15, hi = lane >> 4;
  const int m2 = blockIdx.x >> 3, nb = blockIdx.x & 7;
  const int n_t0 = nb * 16 + (tid >> 5) * 2;    // wave's 2 N-tiles
  const int b0 = m2 * 32;                       // WG's 32 rows (2 M-tiles)
  const unsigned ldsbase = (unsigned)(size_t)(&smemA[0]);
  unsigned epoch = 0;

  for (int t = 0; t < T_IN; ++t) {
    // ---- async stage: rows [b0,b0+32) of [h | x_t] into LDS ----
    for (int c = tid; c < 32 * 68; c += NTH) {
      int row = c / 68, cj = c - row * 68;
      unsigned lds = ldsbase + (unsigned)(row * ST0 * 2) +
                     (unsigned)((cj < 64) ? cj * 16 : 1024 + (cj - 64) * 16);
      const half_t* src = (cj < 64)
          ? h16 + (size_t)(b0 + row) * HID + cj * 8
          : x16 + (size_t)(b0 + row) * (T_IN * 32) + (size_t)t * 32 + (cj - 64) * 8;
      async_cp16(lds, src);
    }
    async_wait0();
    __syncthreads();

    FRESH_ZERO_OFF(woff);
    const half_t* Wh = WhhP + woff;
    const half_t* Wi = WihP + woff;

    v8f acc[4];
#pragma unroll
    for (int i = 0; i < 4; ++i)
#pragma unroll
      for (int r = 0; r < 8; ++r) acc[i][r] = 0.0f;

    for (int kk = 0; kk < KT_H + 1; ++kk) {
      v16h a0 = load_a_frag(smemA + m * ST0 + kk * 32 + hi * 8);
      v16h a1 = load_a_frag(smemA + (16 + m) * ST0 + kk * 32 + hi * 8);
      const half_t* wA;
      const half_t* wB;
      if (kk < KT_H) {
        wA = Wh + ((size_t)n_t0 * KT_H + kk) * 512;
        wB = Wh + ((size_t)(n_t0 + 1) * KT_H + kk) * 512;
      } else {
        wA = Wi + (size_t)n_t0 * 512;
        wB = Wi + (size_t)(n_t0 + 1) * 512;
      }
      v16h bA = *(const v16h*)(wA + lane * 16);
      v16h bB = *(const v16h*)(wB + lane * 16);
      acc[0] = WMMA_F16(a0, bA, acc[0]);
      acc[1] = WMMA_F16(a0, bB, acc[1]);
      acc[2] = WMMA_F16(a1, bA, acc[2]);
      acc[3] = WMMA_F16(a1, bB, acc[3]);
    }
    {
      int col = n_t0 * 16 + m;
#pragma unroll
      for (int r = 0; r < 8; ++r) {
        zbuf[(size_t)(b0 + hi * 8 + r) * G4 + col]           = acc[0][r];
        zbuf[(size_t)(b0 + hi * 8 + r) * G4 + col + 16]      = acc[1][r];
        zbuf[(size_t)(b0 + 16 + hi * 8 + r) * G4 + col]      = acc[2][r];
        zbuf[(size_t)(b0 + 16 + hi * 8 + r) * G4 + col + 16] = acc[3][r];
      }
    }
    grid_barrier(bar, &epoch);

    for (int idx = blockIdx.x * NTH + tid; idx < BATCH * HID; idx += NWG * NTH) {
      int b = idx >> 9, u = idx & (HID - 1);
      const float* z = zbuf + (size_t)b * G4;
      float zi = z[u]           + b_ih[u]           + b_hh[u];
      float zf = z[HID + u]     + b_ih[HID + u]     + b_hh[HID + u];
      float zg = z[2 * HID + u] + b_ih[2 * HID + u] + b_hh[2 * HID + u];
      float zo = z[3 * HID + u] + b_ih[3 * HID + u] + b_hh[3 * HID + u];
      float c = sigm(zf) * cbuf[idx] + sigm(zi) * tanh_(zg);
      float h = sigm(zo) * tanh_(c);
      cbuf[idx] = c;
      half_t hq = (half_t)h;
      h16[idx] = hq;
      ys0[(size_t)t * BATCH * HID + idx] = hq;
    }
    grid_barrier(bar, &epoch);
  }
}

// ---------------- Encoder layer 1: z = h@Whh1^T + ys0_t@Wih1^T ----------------
__global__ __launch_bounds__(NTH)
void enc_layer1(const half_t* __restrict__ ys0,
                const half_t* WhhP, const half_t* WihP,
                const float* __restrict__ b_ih, const float* __restrict__ b_hh,
                float* __restrict__ cbuf, half_t* __restrict__ h16,
                float* __restrict__ zbuf, unsigned* __restrict__ bar) {
  __shared__ half_t smemA[32 * ST1];
  const int tid = threadIdx.x;
  const int lane = tid & 31;
  const int m = lane & 15, hi = lane >> 4;
  const int m2 = blockIdx.x >> 3, nb = blockIdx.x & 7;
  const int n_t0 = nb * 16 + (tid >> 5) * 2;
  const int b0 = m2 * 32;
  const unsigned ldsbase = (unsigned)(size_t)(&smemA[0]);
  unsigned epoch = 0;

  for (int t = 0; t < T_IN; ++t) {
    for (int c = tid; c < 32 * 128; c += NTH) {
      int row = c >> 7, cj = c & 127;
      unsigned lds = ldsbase + (unsigned)(row * ST1 * 2) + (unsigned)(cj * 16);
      const half_t* src = (cj < 64)
          ? h16 + (size_t)(b0 + row) * HID + cj * 8
          : ys0 + ((size_t)t * BATCH + b0 + row) * HID + (cj - 64) * 8;
      async_cp16(lds, src);
    }
    async_wait0();
    __syncthreads();

    FRESH_ZERO_OFF(woff);
    const half_t* Wh = WhhP + woff;
    const half_t* Wi = WihP + woff;

    v8f acc[4];
#pragma unroll
    for (int i = 0; i < 4; ++i)
#pragma unroll
      for (int r = 0; r < 8; ++r) acc[i][r] = 0.0f;

    for (int kk = 0; kk < 2 * KT_H; ++kk) {
      v16h a0 = load_a_frag(smemA + m * ST1 + kk * 32 + hi * 8);
      v16h a1 = load_a_frag(smemA + (16 + m) * ST1 + kk * 32 + hi * 8);
      const half_t* wA;
      const half_t* wB;
      if (kk < KT_H) {
        wA = Wh + ((size_t)n_t0 * KT_H + kk) * 512;
        wB = Wh + ((size_t)(n_t0 + 1) * KT_H + kk) * 512;
      } else {
        wA = Wi + ((size_t)n_t0 * KT_H + (kk - KT_H)) * 512;
        wB = Wi + ((size_t)(n_t0 + 1) * KT_H + (kk - KT_H)) * 512;
      }
      v16h bA = *(const v16h*)(wA + lane * 16);
      v16h bB = *(const v16h*)(wB + lane * 16);
      acc[0] = WMMA_F16(a0, bA, acc[0]);
      acc[1] = WMMA_F16(a0, bB, acc[1]);
      acc[2] = WMMA_F16(a1, bA, acc[2]);
      acc[3] = WMMA_F16(a1, bB, acc[3]);
    }
    {
      int col = n_t0 * 16 + m;
#pragma unroll
      for (int r = 0; r < 8; ++r) {
        zbuf[(size_t)(b0 + hi * 8 + r) * G4 + col]           = acc[0][r];
        zbuf[(size_t)(b0 + hi * 8 + r) * G4 + col + 16]      = acc[1][r];
        zbuf[(size_t)(b0 + 16 + hi * 8 + r) * G4 + col]      = acc[2][r];
        zbuf[(size_t)(b0 + 16 + hi * 8 + r) * G4 + col + 16] = acc[3][r];
      }
    }
    grid_barrier(bar, &epoch);

    for (int idx = blockIdx.x * NTH + tid; idx < BATCH * HID; idx += NWG * NTH) {
      int b = idx >> 9, u = idx & (HID - 1);
      const float* z = zbuf + (size_t)b * G4;
      float zi = z[u]           + b_ih[u]           + b_hh[u];
      float zf = z[HID + u]     + b_ih[HID + u]     + b_hh[HID + u];
      float zg = z[2 * HID + u] + b_ih[2 * HID + u] + b_hh[2 * HID + u];
      float zo = z[3 * HID + u] + b_ih[3 * HID + u] + b_hh[3 * HID + u];
      float c = sigm(zf) * cbuf[idx] + sigm(zi) * tanh_(zg);
      float h = sigm(zo) * tanh_(c);
      cbuf[idx] = c;
      h16[idx] = (half_t)h;
    }
    grid_barrier(bar, &epoch);
  }
}

// ---------------- Decoder: autoregressive 2-layer LSTM + FC ----------------
__global__ __launch_bounds__(NTH)
void decoder_kernel(const half_t* Whh0P, const half_t* Wih1P, const half_t* Whh1P,
                    const float* __restrict__ wih0,   // dW_ih0 [2048][1]
                    const float* __restrict__ bi0, const float* __restrict__ bh0,
                    const float* __restrict__ bi1, const float* __restrict__ bh1,
                    const float* fcwp, const float* __restrict__ fcb,
                    float* __restrict__ c0, half_t* __restrict__ h0q,
                    float* __restrict__ h1, float* __restrict__ c1, half_t* __restrict__ h1q,
                    float* __restrict__ inp,
                    float* __restrict__ zbuf,
                    float* __restrict__ out,
                    unsigned* __restrict__ bar) {
  __shared__ half_t smemA[32 * ST1];
  const int tid = threadIdx.x;
  const int lane = tid & 31;
  const int m = lane & 15, hi = lane >> 4;
  const int m2 = blockIdx.x >> 3, nb = blockIdx.x & 7;
  const int n_t0 = nb * 16 + (tid >> 5) * 2;
  const int b0 = m2 * 32;
  const unsigned ldsbase = (unsigned)(size_t)(&smemA[0]);
  unsigned epoch = 0;

  for (int t = 0; t < T_OUT; ++t) {
    // ======== cell0: z = inp (x) wih0 + h0 @ Whh0^T ========
    for (int c = tid; c < 32 * 64; c += NTH) {
      int row = c >> 6, cj = c & 63;
      unsigned lds = ldsbase + (unsigned)(row * ST1 * 2) + (unsigned)(cj * 16);
      async_cp16(lds, h0q + (size_t)(b0 + row) * HID + cj * 8);
    }
    async_wait0();
    __syncthreads();

    FRESH_ZERO_OFF(woff0);
    const half_t* Wh = Whh0P + woff0;

    v8f acc[4];
    {
      float ivA[8], ivB[8];
#pragma unroll
      for (int r = 0; r < 8; ++r) {
        ivA[r] = inp[b0 + hi * 8 + r];
        ivB[r] = inp[b0 + 16 + hi * 8 + r];
      }
      float wnA = wih0[n_t0 * 16 + m];
      float wnB = wih0[(n_t0 + 1) * 16 + m];
#pragma unroll
      for (int r = 0; r < 8; ++r) {
        acc[0][r] = ivA[r] * wnA;
        acc[1][r] = ivA[r] * wnB;
        acc[2][r] = ivB[r] * wnA;
        acc[3][r] = ivB[r] * wnB;
      }
    }
    for (int kk = 0; kk < KT_H; ++kk) {
      v16h a0 = load_a_frag(smemA + m * ST1 + kk * 32 + hi * 8);
      v16h a1 = load_a_frag(smemA + (16 + m) * ST1 + kk * 32 + hi * 8);
      v16h bA = *(const v16h*)(Wh + ((size_t)n_t0 * KT_H + kk) * 512 + lane * 16);
      v16h bB = *(const v16h*)(Wh + ((size_t)(n_t0 + 1) * KT_H + kk) * 512 + lane * 16);
      acc[0] = WMMA_F16(a0, bA, acc[0]);
      acc[1] = WMMA_F16(a0, bB, acc[1]);
      acc[2] = WMMA_F16(a1, bA, acc[2]);
      acc[3] = WMMA_F16(a1, bB, acc[3]);
    }
    {
      int col = n_t0 * 16 + m;
#pragma unroll
      for (int r = 0; r < 8; ++r) {
        zbuf[(size_t)(b0 + hi * 8 + r) * G4 + col]           = acc[0][r];
        zbuf[(size_t)(b0 + hi * 8 + r) * G4 + col + 16]      = acc[1][r];
        zbuf[(size_t)(b0 + 16 + hi * 8 + r) * G4 + col]      = acc[2][r];
        zbuf[(size_t)(b0 + 16 + hi * 8 + r) * G4 + col + 16] = acc[3][r];
      }
    }
    grid_barrier(bar, &epoch);

    for (int idx = blockIdx.x * NTH + tid; idx < BATCH * HID; idx += NWG * NTH) {
      int b = idx >> 9, u = idx & (HID - 1);
      const float* z = zbuf + (size_t)b * G4;
      float zi = z[u]           + bi0[u]           + bh0[u];
      float zf = z[HID + u]     + bi0[HID + u]     + bh0[HID + u];
      float zg = z[2 * HID + u] + bi0[2 * HID + u] + bh0[2 * HID + u];
      float zo = z[3 * HID + u] + bi0[3 * HID + u] + bh0[3 * HID + u];
      float c = sigm(zf) * c0[idx] + sigm(zi) * tanh_(zg);
      float h = sigm(zo) * tanh_(c);
      c0[idx] = c;
      h0q[idx] = (half_t)h;
    }
    grid_barrier(bar, &epoch);

    // ======== cell1: z = h1 @ Whh1^T + h0_new @ Wih1^T ========
    for (int c = tid; c < 32 * 128; c += NTH) {
      int row = c >> 7, cj = c & 127;
      unsigned lds = ldsbase + (unsigned)(row * ST1 * 2) + (unsigned)(cj * 16);
      const half_t* src = (cj < 64)
          ? h1q + (size_t)(b0 + row) * HID + cj * 8
          : h0q + (size_t)(b0 + row) * HID + (cj - 64) * 8;
      async_cp16(lds, src);
    }
    async_wait0();
    __syncthreads();

    FRESH_ZERO_OFF(woff1);
    const half_t* Wh1 = Whh1P + woff1;
    const half_t* Wi1 = Wih1P + woff1;

#pragma unroll
    for (int i = 0; i < 4; ++i)
#pragma unroll
      for (int r = 0; r < 8; ++r) acc[i][r] = 0.0f;

    for (int kk = 0; kk < 2 * KT_H; ++kk) {
      v16h a0 = load_a_frag(smemA + m * ST1 + kk * 32 + hi * 8);
      v16h a1 = load_a_frag(smemA + (16 + m) * ST1 + kk * 32 + hi * 8);
      const half_t* wA;
      const half_t* wB;
      if (kk < KT_H) {
        wA = Wh1 + ((size_t)n_t0 * KT_H + kk) * 512;
        wB = Wh1 + ((size_t)(n_t0 + 1) * KT_H + kk) * 512;
      } else {
        wA = Wi1 + ((size_t)n_t0 * KT_H + (kk - KT_H)) * 512;
        wB = Wi1 + ((size_t)(n_t0 + 1) * KT_H + (kk - KT_H)) * 512;
      }
      v16h bA = *(const v16h*)(wA + lane * 16);
      v16h bB = *(const v16h*)(wB + lane * 16);
      acc[0] = WMMA_F16(a0, bA, acc[0]);
      acc[1] = WMMA_F16(a0, bB, acc[1]);
      acc[2] = WMMA_F16(a1, bA, acc[2]);
      acc[3] = WMMA_F16(a1, bB, acc[3]);
    }
    {
      int col = n_t0 * 16 + m;
#pragma unroll
      for (int r = 0; r < 8; ++r) {
        zbuf[(size_t)(b0 + hi * 8 + r) * G4 + col]           = acc[0][r];
        zbuf[(size_t)(b0 + hi * 8 + r) * G4 + col + 16]      = acc[1][r];
        zbuf[(size_t)(b0 + 16 + hi * 8 + r) * G4 + col]      = acc[2][r];
        zbuf[(size_t)(b0 + 16 + hi * 8 + r) * G4 + col + 16] = acc[3][r];
      }
    }
    grid_barrier(bar, &epoch);

    for (int idx = blockIdx.x * NTH + tid; idx < BATCH * HID; idx += NWG * NTH) {
      int b = idx >> 9, u = idx & (HID - 1);
      const float* z = zbuf + (size_t)b * G4;
      float zi = z[u]           + bi1[u]           + bh1[u];
      float zf = z[HID + u]     + bi1[HID + u]     + bh1[HID + u];
      float zg = z[2 * HID + u] + bi1[2 * HID + u] + bh1[2 * HID + u];
      float zo = z[3 * HID + u] + bi1[3 * HID + u] + bh1[3 * HID + u];
      float c = sigm(zf) * c1[idx] + sigm(zi) * tanh_(zg);
      float h = sigm(zo) * tanh_(c);
      c1[idx] = c;
      h1[idx] = h;
      h1q[idx] = (half_t)h;
    }
    grid_barrier(bar, &epoch);

    // ======== FC: out[b] = h1[b,:].fcw + fcb ========
    if (blockIdx.x == 0 && tid < BATCH) {
      FRESH_ZERO_OFF(foff);
      const float* fw = fcwp + foff;
      float s = fcb[0];
      const float* hv = h1 + (size_t)tid * HID;
      for (int k = 0; k < HID; ++k) s += hv[k] * fw[k];
      inp[tid] = s;
      out[(size_t)tid * T_OUT + t] = s;
    }
    grid_barrier(bar, &epoch);
  }
}

extern "C" void kernel_launch(void* const* d_in, const int* in_sizes, int n_in,
                              void* d_out, int out_size, void* d_ws, size_t ws_size,
                              hipStream_t stream) {
  (void)in_sizes; (void)n_in; (void)out_size; (void)ws_size;
  const float* x     = (const float*)d_in[0];
  const float* eWih0 = (const float*)d_in[1];
  const float* eWhh0 = (const float*)d_in[2];
  const float* ebih0 = (const float*)d_in[3];
  const float* ebhh0 = (const float*)d_in[4];
  const float* eWih1 = (const float*)d_in[5];
  const float* eWhh1 = (const float*)d_in[6];
  const float* ebih1 = (const float*)d_in[7];
  const float* ebhh1 = (const float*)d_in[8];
  const float* dWih0 = (const float*)d_in[9];
  const float* dWhh0 = (const float*)d_in[10];
  const float* dbih0 = (const float*)d_in[11];
  const float* dbhh0 = (const float*)d_in[12];
  const float* dWih1 = (const float*)d_in[13];
  const float* dWhh1 = (const float*)d_in[14];
  const float* dbih1 = (const float*)d_in[15];
  const float* dbhh1 = (const float*)d_in[16];
  const float* fcw   = (const float*)d_in[17];
  const float* fcb   = (const float*)d_in[18];
  float* out = (float*)d_out;

  char* ws = (char*)d_ws;
  size_t off = 0;
  auto take = [&](size_t bytes) -> char* {
    off = (off + 255) & ~(size_t)255;
    char* p = ws + off;
    off += bytes;
    return p;
  };

  unsigned* bar = (unsigned*)take(256);

  // Contiguous zero-initialized state block.
  const size_t SB = (size_t)BATCH * HID;
  char* states = take(SB * 4 * 3 + SB * 2 * 2 + 4096);
  float* c0   = (float*)states;
  float* c1   = c0 + SB;
  float* h1   = c1 + SB;
  half_t* h0q = (half_t*)(h1 + SB);
  half_t* h1q = h0q + SB;
  float* inp  = (float*)(h1q + SB);
  size_t states_bytes = (size_t)((char*)(inp + BATCH) - states);

  float*  zbuf  = (float*)take((size_t)BATCH * G4 * 4);
  half_t* x16   = (half_t*)take((size_t)BATCH * T_IN * 32 * 2);
  half_t* ys0   = (half_t*)take((size_t)T_IN * BATCH * HID * 2);
  half_t* Whh0P = (half_t*)take((size_t)G4 * HID * 2);
  half_t* Wih0P = (half_t*)take((size_t)G4 * 32 * 2);
  half_t* Wih1P = (half_t*)take((size_t)G4 * HID * 2);
  half_t* Whh1P = (half_t*)take((size_t)G4 * HID * 2);
  half_t* dWhh0P = (half_t*)take((size_t)G4 * HID * 2);
  half_t* dWih1P = (half_t*)take((size_t)G4 * HID * 2);
  half_t* dWhh1P = (half_t*)take((size_t)G4 * HID * 2);

  hipMemsetAsync(bar, 0, 256, stream);
  hipMemsetAsync(states, 0, states_bytes, stream);

  pack_w<<<512, 256, 0, stream>>>(eWhh0, Whh0P, G4, HID);
  pack_w<<<64,  256, 0, stream>>>(eWih0, Wih0P, G4, 32);
  pack_w<<<512, 256, 0, stream>>>(eWih1, Wih1P, G4, HID);
  pack_w<<<512, 256, 0, stream>>>(eWhh1, Whh1P, G4, HID);
  pack_w<<<512, 256, 0, stream>>>(dWhh0, dWhh0P, G4, HID);
  pack_w<<<512, 256, 0, stream>>>(dWih1, dWih1P, G4, HID);
  pack_w<<<512, 256, 0, stream>>>(dWhh1, dWhh1P, G4, HID);
  cvt_h<<<256, 256, 0, stream>>>(x, x16, BATCH * T_IN * 32);

  enc_layer0<<<NWG, NTH, 0, stream>>>(x16, Whh0P, Wih0P, ebih0, ebhh0,
                                      c0, h0q, ys0, zbuf, bar + 0);
  enc_layer1<<<NWG, NTH, 0, stream>>>(ys0, Whh1P, Wih1P, ebih1, ebhh1,
                                      c1, h1q, zbuf, bar + 16);
  decoder_kernel<<<NWG, NTH, 0, stream>>>(dWhh0P, dWih1P, dWhh1P, dWih0,
                                          dbih0, dbhh0, dbih1, dbhh1, fcw, fcb,
                                          c0, h0q, h1, c1, h1q,
                                          inp, zbuf, out, bar + 32);
}
